// RGCN_38680475468491
// MI455X (gfx1250) — compile-verified, matrix operand order checked by reading, not run on hardware
//
#include <hip/hip_runtime.h>
#include <hip/hip_bf16.h>

// RGCN forward (2 layers) for MI455X / gfx1250, wave32 + WMMA f16->f32.
//
// Sizes (fixed by the reference):
#define NN    50000      // nodes
#define EE    800000     // edges
#define DIN   788        // input features
#define KP    800        // DIN padded to multiple of 32 for WMMA K
#define DH    256        // hidden
#define NREL  7
#define NCLS  2

typedef __attribute__((ext_vector_type(8)))  _Float16 v8h;
typedef __attribute__((ext_vector_type(16))) _Float16 v16h;
typedef __attribute__((ext_vector_type(8)))  float    v8f;
typedef __attribute__((ext_vector_type(4)))  int      v4i;

typedef __attribute__((address_space(1))) v4i gv4i;   // global int4
typedef __attribute__((address_space(3))) v4i lv4i;   // LDS int4

#if defined(__has_builtin)
#if __has_builtin(__builtin_amdgcn_global_load_async_to_lds_b128)
#define USE_ASYNC_LDS 1
#endif
#endif

// ---------------------------------------------------------------------------
// f32 -> f16 conversion / padding of node features: xh is (NN, KP), K-padded.
__global__ __launch_bounds__(256) void rgcn_cvt_x(const float* __restrict__ x,
                                                  _Float16* __restrict__ xh) {
  long i = (long)blockIdx.x * 256 + threadIdx.x;
  if (i >= (long)NN * KP) return;
  int row = (int)(i / KP), k = (int)(i % KP);
  float v = (k < DIN) ? x[(size_t)row * DIN + k] : 0.0f;
  xh[i] = (_Float16)v;
}

// Weights -> f16, transposed so each output column is K-contiguous:
// wh layout (8, DH, KP):  wh[r][n][k] = (r<7 ? w1_rel[r][k][n] : w1_root[k][n])
__global__ __launch_bounds__(256) void rgcn_cvt_w(const float* __restrict__ w_rel,
                                                  const float* __restrict__ w_root,
                                                  _Float16* __restrict__ wh) {
  int i = blockIdx.x * 256 + threadIdx.x;      // 8*DH*KP total
  if (i >= 8 * DH * KP) return;
  int r = i / (DH * KP);
  int rem = i % (DH * KP);
  int n = rem / KP, k = rem % KP;
  float v = 0.0f;
  if (k < DIN)
    v = (r < NREL) ? w_rel[((size_t)r * DIN + k) * DH + n]
                   : w_root[(size_t)k * DH + n];
  wh[i] = (_Float16)v;
}

__global__ __launch_bounds__(256) void rgcn_zero(float* __restrict__ p, long n) {
  long i = (long)blockIdx.x * 256 + threadIdx.x;
  if (i < n) p[i] = 0.0f;
}

// Per-(relation, dst) edge counts (float so we can fmaxf/divide like the ref).
__global__ __launch_bounds__(256) void rgcn_count(const int* __restrict__ dst,
                                                  const int* __restrict__ et,
                                                  float* __restrict__ cnt) {
  int e = blockIdx.x * 256 + threadIdx.x;
  if (e >= EE) return;
  atomicAdd(&cnt[(size_t)et[e] * NN + dst[e]], 1.0f);
}

// ---------------------------------------------------------------------------
// C(M x 256) = A(M x KP, f16) * B^T where B is stored (DH x KP), i.e. each
// output column is K-contiguous.  8 waves / block; block stages its 64-column
// B slab (100 KB) in LDS (async global->LDS when available), then each wave
// computes a 16x64 tile: 4 v_wmma_f32_16x16x32_f16 per K-step, A fragment
// software-pipelined one K-step ahead, B fragments from LDS (ds_load_b128).
__global__ __launch_bounds__(256) void rgcn_gemm_f16_wmma(
    const _Float16* __restrict__ A,   // (M, KP)
    const _Float16* __restrict__ B,   // (DH, KP)   B[n][k]
    const float*    __restrict__ bias,// DH or nullptr
    float*          __restrict__ C,   // (M, DH)
    int M) {
  __shared__ _Float16 sB[64 * KP];               // 102,400 B

  const int lane = threadIdx.x & 31;
  const int wave = threadIdx.x >> 5;
  const int n0   = blockIdx.y * 64;

  // ---- stage B tile: 64 cols x KP, 6400 16-byte chunks, 25 per thread ----
  {
    const _Float16* gB = B + (size_t)n0 * KP;
#ifdef USE_ASYNC_LDS
#pragma unroll 1
    for (int i = threadIdx.x; i < 64 * KP / 8; i += 256) {
      __builtin_amdgcn_global_load_async_to_lds_b128(
          (gv4i*)(gB + (size_t)i * 8),
          (lv4i*)(sB + (size_t)i * 8),
          0, 0);
    }
#if __has_builtin(__builtin_amdgcn_s_wait_asynccnt)
    __builtin_amdgcn_s_wait_asynccnt(0);
#else
    asm volatile("s_wait_asynccnt 0x0" ::: "memory");
#endif
#else
#pragma unroll 1
    for (int i = threadIdx.x; i < 64 * KP / 8; i += 256) {
      *(v8h*)(sB + (size_t)i * 8) = *(const v8h*)(gB + (size_t)i * 8);
    }
#endif
    __syncthreads();
  }

  // ---- per-wave 16x64 tile ----
  // Clamp instead of early-exit: keeps EXEC all-ones for WMMA and keeps every
  // wave in the barrier.  Duplicated tiles write identical values (M%16==0).
  const int mtiles = M / 16;
  int mt = blockIdx.x * 8 + wave;
  if (mt >= mtiles) mt = mtiles - 1;

  const int l15 = lane & 15;
  const int kb  = (lane >> 4) << 3;              // 0 or 8, per ISA 16-bit A layout

  const _Float16* ap  = A + (size_t)(mt * 16 + l15) * KP + kb;
  const _Float16* sp0 = sB + (size_t)(l15 +  0) * KP + kb;
  const _Float16* sp1 = sB + (size_t)(l15 + 16) * KP + kb;
  const _Float16* sp2 = sB + (size_t)(l15 + 32) * KP + kb;
  const _Float16* sp3 = sB + (size_t)(l15 + 48) * KP + kb;

  v8f acc0 = {}, acc1 = {}, acc2 = {}, acc3 = {};

  // A fragment pipeline: preload step 0, keep one step in flight.
  v8h al = *(const v8h*)(ap);
  v8h ah = *(const v8h*)(ap + 16);

#pragma unroll 5
  for (int kk = 0; kk < KP; kk += 32) {
    const int kn = (kk + 32 < KP) ? (kk + 32) : 0;   // wrap: redundant but valid
    v8h aln = *(const v8h*)(ap + kn);
    v8h ahn = *(const v8h*)(ap + kn + 16);

    v16h a = __builtin_shufflevector(al, ah, 0,1,2,3,4,5,6,7,8,9,10,11,12,13,14,15);

    v8h b0l = *(const v8h*)(sp0 + kk); v8h b0h = *(const v8h*)(sp0 + kk + 16);
    v8h b1l = *(const v8h*)(sp1 + kk); v8h b1h = *(const v8h*)(sp1 + kk + 16);
    v8h b2l = *(const v8h*)(sp2 + kk); v8h b2h = *(const v8h*)(sp2 + kk + 16);
    v8h b3l = *(const v8h*)(sp3 + kk); v8h b3h = *(const v8h*)(sp3 + kk + 16);
    v16h b0 = __builtin_shufflevector(b0l, b0h, 0,1,2,3,4,5,6,7,8,9,10,11,12,13,14,15);
    v16h b1 = __builtin_shufflevector(b1l, b1h, 0,1,2,3,4,5,6,7,8,9,10,11,12,13,14,15);
    v16h b2 = __builtin_shufflevector(b2l, b2h, 0,1,2,3,4,5,6,7,8,9,10,11,12,13,14,15);
    v16h b3 = __builtin_shufflevector(b3l, b3h, 0,1,2,3,4,5,6,7,8,9,10,11,12,13,14,15);

    acc0 = __builtin_amdgcn_wmma_f32_16x16x32_f16(false, a, false, b0, (short)0, acc0, false, false);
    acc1 = __builtin_amdgcn_wmma_f32_16x16x32_f16(false, a, false, b1, (short)0, acc1, false, false);
    acc2 = __builtin_amdgcn_wmma_f32_16x16x32_f16(false, a, false, b2, (short)0, acc2, false, false);
    acc3 = __builtin_amdgcn_wmma_f32_16x16x32_f16(false, a, false, b3, (short)0, acc3, false, false);

    al = aln; ah = ahn;
  }

  // C/D layout: VGPR r -> M = r + 8*(lane>=16), N = lane&15 (per 16-col tile)
  const int mrow = mt * 16 + ((lane >> 4) << 3);
  float* cp = C + (size_t)mrow * DH + n0 + l15;
  const float bv0 = bias ? bias[n0 + l15 +  0] : 0.0f;
  const float bv1 = bias ? bias[n0 + l15 + 16] : 0.0f;
  const float bv2 = bias ? bias[n0 + l15 + 32] : 0.0f;
  const float bv3 = bias ? bias[n0 + l15 + 48] : 0.0f;
#pragma unroll
  for (int r = 0; r < 8; ++r) {
    cp[(size_t)r * DH +  0] = acc0[r] + bv0;
    cp[(size_t)r * DH + 16] = acc1[r] + bv1;
    cp[(size_t)r * DH + 32] = acc2[r] + bv2;
    cp[(size_t)r * DH + 48] = acc3[r] + bv3;
  }
}

// ---------------------------------------------------------------------------
// Layer-1 scatter-mean for one relation: h[dst] += x_rel[src] / cnt[rel,dst].
// One block per edge, 256 threads = 256 features.
__global__ __launch_bounds__(256) void rgcn_scatter1(const int* __restrict__ src,
                                                     const int* __restrict__ dst,
                                                     const int* __restrict__ et,
                                                     const float* __restrict__ cnt,
                                                     const float* __restrict__ xr,
                                                     float* __restrict__ h,
                                                     int rel) {
  int e = blockIdx.x;
  if (et[e] != rel) return;
  int s = src[e], d = dst[e];
  float inv = 1.0f / fmaxf(cnt[(size_t)rel * NN + d], 1.0f);
  int f = threadIdx.x;
  atomicAdd(&h[(size_t)d * DH + f], xr[(size_t)s * DH + f] * inv);
}

// ---------------------------------------------------------------------------
// Layer 2 node-local part: relu(h) then 16 dot products of length 256:
//   r<7 -> hr2[r][node][c] = relu(h[node]) . w2_rel[r][:,c]
//   r=7 -> out[node][c]    = relu(h[node]) . w2_root[:,c] + b2[c]
__global__ __launch_bounds__(256) void rgcn_layer2(const float* __restrict__ h,
                                                   const float* __restrict__ w2_rel,
                                                   const float* __restrict__ w2_root,
                                                   const float* __restrict__ b2,
                                                   float* __restrict__ out,
                                                   float* __restrict__ hr2) {
  __shared__ float sh[DH];
  int node = blockIdx.x;
  int t = threadIdx.x;
  float v = h[(size_t)node * DH + t];
  sh[t] = v > 0.0f ? v : 0.0f;
  __syncthreads();
  if (t < 16) {
    int r = t >> 1, c = t & 1;
    float s = 0.0f;
    if (r < NREL) {
      const float* w = w2_rel + (size_t)r * DH * NCLS + c;
#pragma unroll 8
      for (int f = 0; f < DH; ++f) s += sh[f] * w[f * NCLS];
      hr2[((size_t)r * NN + node) * NCLS + c] = s;
    } else {
      const float* w = w2_root + c;
#pragma unroll 8
      for (int f = 0; f < DH; ++f) s += sh[f] * w[f * NCLS];
      out[(size_t)node * NCLS + c] = s + b2[c];
    }
  }
}

// Layer-2 scatter-mean: out[dst][c] += hr2[rel][src][c] / cnt[rel,dst].
__global__ __launch_bounds__(256) void rgcn_scatter2(const int* __restrict__ src,
                                                     const int* __restrict__ dst,
                                                     const int* __restrict__ et,
                                                     const float* __restrict__ cnt,
                                                     const float* __restrict__ hr2,
                                                     float* __restrict__ out) {
  int e = blockIdx.x * 256 + threadIdx.x;
  if (e >= EE) return;
  int r = et[e], s = src[e], d = dst[e];
  float inv = 1.0f / fmaxf(cnt[(size_t)r * NN + d], 1.0f);
  const float* hp = hr2 + ((size_t)r * NN + s) * NCLS;
  atomicAdd(&out[(size_t)d * NCLS + 0], hp[0] * inv);
  atomicAdd(&out[(size_t)d * NCLS + 1], hp[1] * inv);
}

// ---------------------------------------------------------------------------
extern "C" void kernel_launch(void* const* d_in, const int* in_sizes, int n_in,
                              void* d_out, int out_size, void* d_ws, size_t ws_size,
                              hipStream_t stream) {
  (void)in_sizes; (void)n_in; (void)out_size; (void)ws_size;

  const float* x       = (const float*)d_in[0];          // (NN, DIN)
  const int*   ei      = (const int*)d_in[1];            // (2, EE): [0]=src, [1]=dst
  const int*   et      = (const int*)d_in[2];            // (EE,)
  const float* w1_rel  = (const float*)d_in[3];          // (7, DIN, DH)
  const float* w1_root = (const float*)d_in[4];          // (DIN, DH)
  const float* b1      = (const float*)d_in[5];          // (DH,)
  const float* w2_rel  = (const float*)d_in[6];          // (7, DH, NCLS)
  const float* w2_root = (const float*)d_in[7];          // (DH, NCLS)
  const float* b2      = (const float*)d_in[8];          // (NCLS,)
  float*       out     = (float*)d_out;                  // (NN, NCLS)

  const int* src = ei;
  const int* dst = ei + EE;

  // Workspace layout (all 16B-aligned offsets), total ~190 MB:
  char* ws = (char*)d_ws;
  _Float16* xh  = (_Float16*)(ws);                       //  80,000,000 B  (NN x KP f16)
  _Float16* wh  = (_Float16*)(ws +  80000000);           //   3,276,800 B  (8 x DH x KP f16)
  float*    xr  = (float*)   (ws +  83276800);           //  51,200,000 B  (NN x DH, reused per rel)
  float*    h   = (float*)   (ws + 134476800);           //  51,200,000 B  (NN x DH)
  float*    cnt = (float*)   (ws + 185676800);           //   1,400,000 B  (7 x NN)
  float*    hr2 = (float*)   (ws + 187076800);           //   2,800,000 B  (7 x NN x NCLS)

  // 1) f16 staging (x fits in the 192MB L2 after this: GEMMs are L2-resident).
  rgcn_cvt_x<<<(int)(((long)NN * KP + 255) / 256), 256, 0, stream>>>(x, xh);
  rgcn_cvt_w<<<(8 * DH * KP + 255) / 256, 256, 0, stream>>>(w1_rel, w1_root, wh);

  // 2) per-(rel,dst) counts.
  rgcn_zero<<<(NREL * NN + 255) / 256, 256, 0, stream>>>(cnt, (long)NREL * NN);
  rgcn_count<<<(EE + 255) / 256, 256, 0, stream>>>(dst, et, cnt);

  // 3) root transform: h = x @ w1_root + b1  (WMMA GEMM).
  dim3 ggrid((NN / 16 + 7) / 8, DH / 64);   // (391, 4)
  rgcn_gemm_f16_wmma<<<ggrid, 256, 0, stream>>>(xh, wh + (size_t)NREL * DH * KP, b1, h, NN);

  // 4) per-relation transform + scatter-mean into h (single reused xr buffer).
  for (int r = 0; r < NREL; ++r) {
    rgcn_gemm_f16_wmma<<<ggrid, 256, 0, stream>>>(xh, wh + (size_t)r * DH * KP, nullptr, xr, NN);
    rgcn_scatter1<<<EE, 256, 0, stream>>>(src, dst, et, cnt, xr, h, r);
  }

  // 5) layer 2: relu + tiny per-node GEMV (root part -> out, rel parts -> hr2),
  //    then edge scatter-mean into out.
  rgcn_layer2<<<NN, 256, 0, stream>>>(h, w2_rel, w2_root, b2, out, hr2);
  rgcn_scatter2<<<(EE + 255) / 256, 256, 0, stream>>>(src, dst, et, cnt, hr2, out);
}
